// BinarizedConvNet_74148315398250
// MI455X (gfx1250) — matrix operand order repaired and checked by reading, not compiled, not run on hardware
//
#include <hip/hip_runtime.h>

// ---------------------------------------------------------------------------
// Types
// ---------------------------------------------------------------------------
typedef __bf16          v16bf __attribute__((ext_vector_type(16)));
typedef float           v8f   __attribute__((ext_vector_type(8)));
typedef unsigned short  v16us __attribute__((ext_vector_type(16)));
typedef unsigned short  v8us  __attribute__((ext_vector_type(8)));

#define CONV_EPS 1e-5f

// bf16 helpers (round-to-nearest-even), activations stored as u16 bit patterns
__device__ __forceinline__ unsigned short f2bf(float f) {
  unsigned u = __builtin_bit_cast(unsigned, f);
  unsigned r = u + 0x7FFFu + ((u >> 16) & 1u);
  return (unsigned short)(r >> 16);
}
__device__ __forceinline__ float bf2f(unsigned short h) {
  return __builtin_bit_cast(float, (unsigned)h << 16);
}

// ---------------------------------------------------------------------------
// CDNA5 Tensor Data Mover staging (guarded); fallback = cooperative copy
// ---------------------------------------------------------------------------
#if defined(__has_builtin)
#if __has_builtin(__builtin_amdgcn_tensor_load_to_lds) && __has_builtin(__builtin_amdgcn_s_wait_tensorcnt)
#define HAS_TDM 1
#endif
#endif
#ifndef HAS_TDM
#define HAS_TDM 0
#endif

#if HAS_TDM
typedef unsigned int u32x4 __attribute__((ext_vector_type(4)));
typedef int          i32x4 __attribute__((ext_vector_type(4)));
typedef int          i32x8 __attribute__((ext_vector_type(8)));

// 1-D TDM copy: 4096 bytes global -> LDS (data_size=8B, tile_dim0=512)
__device__ __forceinline__ void tdm_copy4k(const void* gsrc, void* ldst) {
  unsigned long long ga = (unsigned long long)gsrc;
  unsigned lds = (unsigned)(unsigned long long)ldst;  // low 32 bits of flat LDS addr = LDS offset
  u32x4 g0;
  g0[0] = 1u;                                           // count=1, user descriptor
  g0[1] = lds;                                          // lds_addr
  g0[2] = (unsigned)ga;                                 // global_addr[31:0]
  g0[3] = (unsigned)((ga >> 32) & 0x01FFFFFFu) | (2u << 30);  // global_addr[56:32] | type=2
  i32x8 g1;
  g1[0] = (int)(3u << 16);            // workgroup_mask=0, data_size=3 (8B)
  g1[1] = (int)(512u << 16);          // tensor_dim0[15:0]
  g1[2] = (int)(0u | (1u << 16));     // tensor_dim0[31:16] | tensor_dim1[15:0]=1
  g1[3] = (int)(0u | (512u << 16));   // tensor_dim1[31:16] | tile_dim0=512
  g1[4] = (int)(1u);                  // tile_dim1=1, tile_dim2=0
  g1[5] = (int)(512u);                // tensor_dim0_stride[31:0]
  g1[6] = (int)(0u | (512u << 16));   // stride0[47:32] | stride1[15:0]
  g1[7] = 0;                          // stride1[47:16]
  i32x4 g2 = {0, 0, 0, 0};
  i32x4 g3 = {0, 0, 0, 0};
#if defined(__clang_major__) && __clang_major__ >= 23
  i32x8 g4 = {0, 0, 0, 0, 0, 0, 0, 0};
  __builtin_amdgcn_tensor_load_to_lds(g0, g1, g2, g3, g4, 0);
#else
  __builtin_amdgcn_tensor_load_to_lds(g0, g1, g2, g3, 0);
#endif
}
#endif

// Stage one 4KB packed-B block (4 ntiles x 32 lanes x 16 bf16) into LDS
__device__ __forceinline__ void stageB(const unsigned short* src, unsigned short* dstLds,
                                       int tid, int wv) {
#if HAS_TDM
  (void)tid;
  if (wv == 0) tdm_copy4k(src, dstLds);
#else
  (void)wv;
  *(v8us*)(dstLds + tid * 8) = *(const v8us*)(src + tid * 8);
#endif
}
__device__ __forceinline__ void stage_wait(int wv) {
#if HAS_TDM
  if (wv == 0) __builtin_amdgcn_s_wait_tensorcnt(0);
#else
  (void)wv;
#endif
  __syncthreads();
}

// ---------------------------------------------------------------------------
// Conv1: Ci=3 (K=27, not WMMA-friendly) -> direct VALU conv, NCHW in, NHWC f32 out
// ---------------------------------------------------------------------------
__global__ __launch_bounds__(256) void k_conv1(const float* __restrict__ x,
                                               const float* __restrict__ w,
                                               const float* __restrict__ b,
                                               float* __restrict__ out) {
  long long idx = (long long)blockIdx.x * 256 + threadIdx.x;  // [0, 262144*128)
  if (idx >= (long long)262144 * 128) return;
  int co = (int)(idx & 127);
  int m  = (int)(idx >> 7);
  int n = m >> 10;              // H*W = 1024
  int rem = m & 1023;
  int h = rem >> 5, wp = rem & 31;
  float s = b[co];
#pragma unroll
  for (int ci = 0; ci < 3; ++ci)
#pragma unroll
    for (int dy = 0; dy < 3; ++dy)
#pragma unroll
      for (int dx = 0; dx < 3; ++dx) {
        int iy = h + dy - 1, ix = wp + dx - 1;
        if (iy < 0 || iy > 31 || ix < 0 || ix > 31) continue;
        float xv = x[((n * 3 + ci) * 32 + iy) * 32 + ix];
        float wv = w[((co * 3 + ci) * 3 + dy) * 3 + dx];
        s += (wv >= 0.f) ? xv : -xv;  // binarized weight
      }
  out[(size_t)m * 128 + co] = s;
}

// ---------------------------------------------------------------------------
// BatchNorm (training-mode) stats: one block per channel over NHWC f32
// ---------------------------------------------------------------------------
__global__ __launch_bounds__(256) void k_bn_stats(const float* __restrict__ xin,
                                                  float* __restrict__ stats, int M, int C) {
  __shared__ float ssum[256];
  __shared__ float ssq[256];
  int c = blockIdx.x, t = threadIdx.x;
  float s = 0.f, q = 0.f;
  for (int i = t; i < M; i += 256) {
    float v = xin[(size_t)i * C + c];
    s += v; q += v * v;
  }
  ssum[t] = s; ssq[t] = q;
  __syncthreads();
  for (int st = 128; st > 0; st >>= 1) {
    if (t < st) { ssum[t] += ssum[t + st]; ssq[t] += ssq[t + st]; }
    __syncthreads();
  }
  if (t == 0) {
    float mean = ssum[0] / (float)M;
    float var  = ssq[0] / (float)M - mean * mean;  // ddof=0, matches jnp.var
    stats[c]     = mean;
    stats[C + c] = rsqrtf(var + CONV_EPS);
  }
}

__global__ __launch_bounds__(256) void k_bn_relu(const float* __restrict__ xin,
                                                 const float* __restrict__ stats,
                                                 const float* __restrict__ gg,
                                                 const float* __restrict__ bb,
                                                 unsigned short* __restrict__ outp,
                                                 long long total, int C) {
  long long idx = (long long)blockIdx.x * 256 + threadIdx.x;
  if (idx >= total) return;
  int c = (int)(idx % C);
  float v = (xin[idx] - stats[c]) * stats[C + c] * gg[c] + bb[c];
  outp[idx] = f2bf(fmaxf(v, 0.f));
}

// ---------------------------------------------------------------------------
// 2x2 max-pool on bf16 NHWC
// ---------------------------------------------------------------------------
__global__ __launch_bounds__(256) void k_pool(const unsigned short* __restrict__ in,
                                              unsigned short* __restrict__ outp,
                                              int Nn, int H, int W, int C) {
  int Ho = H >> 1, Wo = W >> 1;
  long long tot = (long long)Nn * Ho * Wo * C;
  long long idx = (long long)blockIdx.x * 256 + threadIdx.x;
  if (idx >= tot) return;
  int c = (int)(idx % C);
  long long r = idx / C;
  int wo = (int)(r % Wo); r /= Wo;
  int ho = (int)(r % Ho);
  int n  = (int)(r / Ho);
  const unsigned short* base = in + (((size_t)(n * H + 2 * ho) * W + 2 * wo) * C + c);
  size_t rs = (size_t)W * C;
  float a = bf2f(base[0]), b = bf2f(base[C]);
  float d = bf2f(base[rs]), e = bf2f(base[rs + C]);
  outp[idx] = f2bf(fmaxf(fmaxf(a, b), fmaxf(d, e)));
}

// ---------------------------------------------------------------------------
// Weight packers: binarize -> bf16 +-1, laid out in per-lane WMMA B-fragment order
// packed[((ktile*(Co/16)+ntile)*32+lane)*16+j]; k_local = (lane>>4)*16 + j; col = lane&15
// ---------------------------------------------------------------------------
__global__ __launch_bounds__(256) void k_pack_conv(const float* __restrict__ cw,
                                                   unsigned short* __restrict__ packed,
                                                   int Ci, int Co) {
  long long tot = (long long)9 * Ci * Co;
  long long idx = (long long)blockIdx.x * 256 + threadIdx.x;
  if (idx >= tot) return;
  int j    = (int)(idx & 15);
  int lane = (int)((idx >> 4) & 31);
  long long rest = idx >> 9;
  int ntiles = Co >> 4;
  int ntile = (int)(rest % ntiles);
  int ktile = (int)(rest / ntiles);
  int k  = ktile * 32 + ((lane >> 4) << 4) + j;   // K index = (dy*3+dx)*Ci + ci
  int co = ntile * 16 + (lane & 15);
  int tt = k / Ci, ci = k % Ci;
  int dy = tt / 3, dx = tt % 3;
  float w = cw[((co * Ci + ci) * 3 + dy) * 3 + dx];
  packed[idx] = (w >= 0.f) ? (unsigned short)0x3F80u : (unsigned short)0xBF80u;
}

__global__ __launch_bounds__(256) void k_pack_fc(const float* __restrict__ fw,
                                                 unsigned short* __restrict__ packed,
                                                 int K, int Nc) {
  long long tot = (long long)K * Nc;
  long long idx = (long long)blockIdx.x * 256 + threadIdx.x;
  if (idx >= tot) return;
  int j    = (int)(idx & 15);
  int lane = (int)((idx >> 4) & 31);
  long long rest = idx >> 9;
  int ntiles = Nc >> 4;
  int ntile = (int)(rest % ntiles);
  int ktile = (int)(rest / ntiles);
  int k = ktile * 32 + ((lane >> 4) << 4) + j;
  int o = ntile * 16 + (lane & 15);
  float w = fw[(size_t)o * K + k];
  packed[idx] = (w >= 0.f) ? (unsigned short)0x3F80u : (unsigned short)0xBF80u;
}

// ---------------------------------------------------------------------------
// Implicit-GEMM conv (3x3, pad 1) via v_wmma_f32_16x16x32_bf16.
// Compile-time H,W,Ci (all powers of two) -> all index math is shifts/masks.
// Per-wave tile: 32(M) x 64(Co) = 2 A-fragments x 4 B-tiles = 8 WMMA / K-step.
// Block tile: 256(M) x 64(Co), 8 waves. B double-buffered in LDS (TDM or coop).
// ---------------------------------------------------------------------------
template <int H, int W, int Ci>
__global__ __launch_bounds__(256) void k_conv_wmma(
    const unsigned short* __restrict__ act,    // [N,H,W,Ci] bf16
    const unsigned short* __restrict__ packB,  // [(9Ci/32)*(Co/16)*512]
    const float* __restrict__ bias,            // [Co]
    float* __restrict__ out,                   // [N,H,W,Co] f32
    int Co) {
  constexpr int HW     = H * W;
  constexpr int ksteps = (9 * Ci) >> 5;

  const int tid  = threadIdx.x;
  const int lane = tid & 31;
  const int wv   = tid >> 5;
  const int col  = lane & 15;
  const int lhi  = lane >> 4;

  const int m0  = blockIdx.x * 256 + wv * 32;   // two 16-row subtiles
  const int co0 = blockIdx.y * 64;
  const int ntiles_total = Co >> 4;
  const int nt0 = co0 >> 4;

  // decode output pixel for each subtile's A-row (all shifts: HW, W are pow2)
  int n_[2], oh_[2], ow_[2];
#pragma unroll
  for (int s = 0; s < 2; ++s) {
    int m   = m0 + s * 16 + col;
    n_[s]   = m / HW;
    int rem = m & (HW - 1);
    oh_[s]  = rem / W;
    ow_[s]  = rem & (W - 1);
  }

  __shared__ alignas(16) unsigned short Bst[2][2048];  // 2 x 4KB

  v8f acc[2][4];
#pragma unroll
  for (int s = 0; s < 2; ++s)
#pragma unroll
    for (int j = 0; j < 4; ++j)
      acc[s][j] = (v8f){0.f, 0.f, 0.f, 0.f, 0.f, 0.f, 0.f, 0.f};

  const v8us zv = {0, 0, 0, 0, 0, 0, 0, 0};

  stageB(packB + ((size_t)nt0 << 9), Bst[0], tid, wv);
  stage_wait(wv);

  int cur = 0;
  for (int kt = 0; kt < ksteps; ++kt) {
    int nxt = cur ^ 1;
    if (kt + 1 < ksteps)
      stageB(packB + ((size_t)((size_t)(kt + 1) * ntiles_total + nt0) << 9), Bst[nxt], tid, wv);

    // ---- tap + channel offset (compile-time divisors) ----
    int kk  = kt << 5;
    int tt  = kk / Ci;             // constexpr divisor -> shift
    int ci0 = kk & (Ci - 1);
    int dy = tt / 3, dx = tt - dy * 3;

    // ---- A fragments (implicit GEMM gather, bounds = zero padding) ----
    v16bf a[2];
#pragma unroll
    for (int s = 0; s < 2; ++s) {
      int iy = oh_[s] + dy - 1, ix = ow_[s] + dx - 1;
      v8us c0 = zv, c1 = zv;
      if (iy >= 0 && iy < H && ix >= 0 && ix < W) {
        const unsigned short* p = act + ((size_t)((n_[s] * H + iy) * W + ix)) * Ci + ci0;
        c0 = *(const v8us*)(p + (lhi ? 16 : 0));
        c1 = *(const v8us*)(p + (lhi ? 24 : 8));
        __builtin_prefetch(p + 32, 0, 0);
      }
      v16us au = __builtin_shufflevector(c0, c1, 0, 1, 2, 3, 4, 5, 6, 7,
                                         8, 9, 10, 11, 12, 13, 14, 15);
      a[s] = __builtin_bit_cast(v16bf, au);
    }

    // ---- 8 WMMAs against LDS-resident B fragments ----
#pragma unroll
    for (int j = 0; j < 4; ++j) {
      v16us bu = *(const v16us*)(&Bst[cur][(j * 32 + lane) * 16]);
      v16bf b = __builtin_bit_cast(v16bf, bu);
      acc[0][j] = __builtin_amdgcn_wmma_f32_16x16x32_bf16(false, a[0], false, b,
                                                          (short)0, acc[0][j], false, false);
      acc[1][j] = __builtin_amdgcn_wmma_f32_16x16x32_bf16(false, a[1], false, b,
                                                          (short)0, acc[1][j], false, false);
    }
    stage_wait(wv);
    cur = nxt;
  }

  // ---- epilogue: C/D layout -> NHWC f32 (+conv bias) ----
#pragma unroll
  for (int j = 0; j < 4; ++j) {
    int co = co0 + j * 16 + col;
    float bv = bias[co];
#pragma unroll
    for (int s = 0; s < 2; ++s)
#pragma unroll
      for (int r = 0; r < 8; ++r) {
        int mrow = m0 + s * 16 + r + lhi * 8;
        out[(size_t)mrow * Co + co] = acc[s][j][r] + bv;
      }
  }
}

// ---------------------------------------------------------------------------
// WMMA GEMM with fused bias+ReLU+bf16 epilogue (FC1/FC2, binarized weights)
// Per-wave 32(M) x 64(N); block 256(M) x 64(N).
// ---------------------------------------------------------------------------
__global__ __launch_bounds__(256) void k_gemm_relu_wmma(
    const unsigned short* __restrict__ A,      // [M,K] bf16
    const unsigned short* __restrict__ packB,  // packed fragments [K/32][Nc/16][512]
    const float* __restrict__ bias,            // [Nc]
    unsigned short* __restrict__ outp,         // [M,Nc] bf16
    int Nc, int K) {
  const int tid  = threadIdx.x;
  const int lane = tid & 31;
  const int wv   = tid >> 5;
  const int col  = lane & 15;
  const int lhi  = lane >> 4;

  const int m0  = blockIdx.x * 256 + wv * 32;
  const int no0 = blockIdx.y * 64;
  const int ntiles_total = Nc >> 4;
  const int nt0 = no0 >> 4;
  const int ksteps = K >> 5;

  __shared__ alignas(16) unsigned short Bst[2][2048];

  v8f acc[2][4];
#pragma unroll
  for (int s = 0; s < 2; ++s)
#pragma unroll
    for (int j = 0; j < 4; ++j)
      acc[s][j] = (v8f){0.f, 0.f, 0.f, 0.f, 0.f, 0.f, 0.f, 0.f};

  const unsigned short* arow[2];
  arow[0] = A + (size_t)(m0 + col) * K;
  arow[1] = A + (size_t)(m0 + 16 + col) * K;

  stageB(packB + ((size_t)nt0 << 9), Bst[0], tid, wv);
  stage_wait(wv);

  int cur = 0;
  for (int kt = 0; kt < ksteps; ++kt) {
    int nxt = cur ^ 1;
    if (kt + 1 < ksteps)
      stageB(packB + ((size_t)((size_t)(kt + 1) * ntiles_total + nt0) << 9), Bst[nxt], tid, wv);

    v16bf a[2];
#pragma unroll
    for (int s = 0; s < 2; ++s) {
      const unsigned short* p = arow[s] + (kt << 5);
      v8us c0 = *(const v8us*)(p + (lhi ? 16 : 0));
      v8us c1 = *(const v8us*)(p + (lhi ? 24 : 8));
      __builtin_prefetch(p + 32, 0, 0);
      v16us au = __builtin_shufflevector(c0, c1, 0, 1, 2, 3, 4, 5, 6, 7,
                                         8, 9, 10, 11, 12, 13, 14, 15);
      a[s] = __builtin_bit_cast(v16bf, au);
    }

#pragma unroll
    for (int j = 0; j < 4; ++j) {
      v16us bu = *(const v16us*)(&Bst[cur][(j * 32 + lane) * 16]);
      v16bf b = __builtin_bit_cast(v16bf, bu);
      acc[0][j] = __builtin_amdgcn_wmma_f32_16x16x32_bf16(false, a[0], false, b,
                                                          (short)0, acc[0][j], false, false);
      acc[1][j] = __builtin_amdgcn_wmma_f32_16x16x32_bf16(false, a[1], false, b,
                                                          (short)0, acc[1][j], false, false);
    }
    stage_wait(wv);
    cur = nxt;
  }

#pragma unroll
  for (int j = 0; j < 4; ++j) {
    int o = no0 + j * 16 + col;
    float bv = bias[o];
#pragma unroll
    for (int s = 0; s < 2; ++s)
#pragma unroll
      for (int r = 0; r < 8; ++r) {
        int mrow = m0 + s * 16 + r + lhi * 8;
        outp[(size_t)mrow * Nc + o] = f2bf(fmaxf(acc[s][j][r] + bv, 0.f));
      }
  }
}

// ---------------------------------------------------------------------------
// NHWC [256,4,4,512] bf16 -> NCHW-flatten [256, 8192] (matches torch .view order)
// ---------------------------------------------------------------------------
__global__ __launch_bounds__(256) void k_flatten(const unsigned short* __restrict__ act,
                                                 unsigned short* __restrict__ outp) {
  int idx = blockIdx.x * 256 + threadIdx.x;  // 256*8192
  if (idx >= 256 * 8192) return;
  int n = idx >> 13;
  int f = idx & 8191;
  int c = f >> 4;
  int r = f & 15;
  int h = r >> 2, w = r & 3;
  outp[idx] = act[((size_t)(n * 4 + h) * 4 + w) * 512 + c];
}

// ---------------------------------------------------------------------------
// FC3: full-precision f32 [256,1024] x [10,1024]^T + b
// ---------------------------------------------------------------------------
__global__ __launch_bounds__(256) void k_fc3(const unsigned short* __restrict__ a,
                                             const float* __restrict__ w,
                                             const float* __restrict__ b,
                                             float* __restrict__ outp) {
  int idx = blockIdx.x * 256 + threadIdx.x;
  if (idx >= 2560) return;
  int o = idx % 10, n = idx / 10;
  float s = b[o];
  for (int k = 0; k < 1024; ++k) s += bf2f(a[n * 1024 + k]) * w[o * 1024 + k];
  outp[idx] = s;
}

// ---------------------------------------------------------------------------
// Host launch sequence
// ---------------------------------------------------------------------------
static inline long long cdiv(long long a, long long b) { return (a + b - 1) / b; }

template <int H, int W, int Ci>
static void launch_conv(const unsigned short* in, const float* cwp, const float* cbp,
                        int Co, unsigned short* packed, float* convF, hipStream_t stream) {
  long long pk = (long long)9 * Ci * Co;
  k_pack_conv<<<(unsigned)cdiv(pk, 256), 256, 0, stream>>>(cwp, packed, Ci, Co);
  int M = 256 * H * W;
  dim3 gr((unsigned)(M / 256), (unsigned)(Co / 64));
  k_conv_wmma<H, W, Ci><<<gr, 256, 0, stream>>>(in, packed, cbp, convF, Co);
}

extern "C" void kernel_launch(void* const* d_in, const int* in_sizes, int n_in,
                              void* d_out, int out_size, void* d_ws, size_t ws_size,
                              hipStream_t stream) {
  (void)in_sizes; (void)n_in; (void)out_size; (void)ws_size;
  const float* x = (const float*)d_in[0];
  const float *cw[7], *cb[7], *g[7], *bt[7];
  for (int i = 1; i <= 6; ++i) {
    int b = 1 + 4 * (i - 1);
    cw[i] = (const float*)d_in[b];     cb[i] = (const float*)d_in[b + 1];
    g[i]  = (const float*)d_in[b + 2]; bt[i] = (const float*)d_in[b + 3];
  }
  const float* fw1 = (const float*)d_in[25]; const float* fb1 = (const float*)d_in[26];
  const float* fw2 = (const float*)d_in[27]; const float* fb2 = (const float*)d_in[28];
  const float* fw3 = (const float*)d_in[29]; const float* fb3 = (const float*)d_in[30];

  char* ws = (char*)d_ws;
  size_t off = 0;
  auto take = [&](size_t bytes) -> void* {
    void* p = ws + off;
    off += (bytes + 255) & ~(size_t)255;
    return p;
  };
  float*          convF  = (float*)take((size_t)262144 * 128 * 4);          // 134 MB
  unsigned short* actA   = (unsigned short*)take((size_t)262144 * 128 * 2); // 67 MB
  unsigned short* actB   = (unsigned short*)take((size_t)262144 * 128 * 2); // 67 MB
  unsigned short* packed = (unsigned short*)take((size_t)8192 * 1024 * 2);  // 16.8 MB
  float*          stats  = (float*)take(4096);

  auto run_bn = [&](unsigned short* outAct, const float* gg, const float* bb, int M, int C) {
    k_bn_stats<<<C, 256, 0, stream>>>(convF, stats, M, C);
    long long tot = (long long)M * C;
    k_bn_relu<<<(unsigned)cdiv(tot, 256), 256, 0, stream>>>(convF, stats, gg, bb, outAct, tot, C);
  };
  auto run_pool = [&](const unsigned short* in, unsigned short* out, int Hh, int Ww, int C) {
    long long tot = (long long)256 * (Hh / 2) * (Ww / 2) * C;
    k_pool<<<(unsigned)cdiv(tot, 256), 256, 0, stream>>>(in, out, 256, Hh, Ww, C);
  };

  // ---- Layer 1 (direct) ----
  k_conv1<<<(unsigned)cdiv((long long)262144 * 128, 256), 256, 0, stream>>>(x, cw[1], cb[1], convF);
  run_bn(actA, g[1], bt[1], 262144, 128);
  // ---- Layer 2 ----
  launch_conv<32, 32, 128>(actA, cw[2], cb[2], 128, packed, convF, stream);
  run_bn(actB, g[2], bt[2], 262144, 128);
  run_pool(actB, actA, 32, 32, 128);            // -> [256,16,16,128]
  // ---- Layer 3 ----
  launch_conv<16, 16, 128>(actA, cw[3], cb[3], 256, packed, convF, stream);
  run_bn(actB, g[3], bt[3], 65536, 256);
  // ---- Layer 4 ----
  launch_conv<16, 16, 256>(actB, cw[4], cb[4], 256, packed, convF, stream);
  run_bn(actA, g[4], bt[4], 65536, 256);
  run_pool(actA, actB, 16, 16, 256);            // -> [256,8,8,256]
  // ---- Layer 5 ----
  launch_conv<8, 8, 256>(actB, cw[5], cb[5], 512, packed, convF, stream);
  run_bn(actA, g[5], bt[5], 16384, 512);
  // ---- Layer 6 ----
  launch_conv<8, 8, 512>(actA, cw[6], cb[6], 512, packed, convF, stream);
  run_bn(actB, g[6], bt[6], 16384, 512);
  run_pool(actB, actA, 8, 8, 512);              // -> [256,4,4,512]
  // ---- Flatten (NCHW order) ----
  k_flatten<<<(unsigned)cdiv(256 * 8192, 256), 256, 0, stream>>>(actA, actB);
  // ---- FC1 ----
  k_pack_fc<<<(unsigned)cdiv((long long)8192 * 1024, 256), 256, 0, stream>>>(fw1, packed, 8192, 1024);
  {
    dim3 gr(1, 16);
    k_gemm_relu_wmma<<<gr, 256, 0, stream>>>(actB, packed, fb1, actA, 1024, 8192);
  }
  // ---- FC2 ----
  k_pack_fc<<<(unsigned)cdiv((long long)1024 * 1024, 256), 256, 0, stream>>>(fw2, packed, 1024, 1024);
  {
    dim3 gr(1, 16);
    k_gemm_relu_wmma<<<gr, 256, 0, stream>>>(actA, packed, fb2, actB, 1024, 1024);
  }
  // ---- FC3 (full precision) ----
  k_fc3<<<(unsigned)cdiv(2560, 256), 256, 0, stream>>>(actB, fw3, fb3, (float*)d_out);
}
